// NNNModel_55817394979041
// MI455X (gfx1250) — matrix operand order repaired and checked by reading, not compiled, not run on hardware
//
#include <hip/hip_runtime.h>
#include <math.h>

// Problem constants (from reference): B=32, T=2048, IN=256, H=512, 4H=2048, C=64
#define BB   32
#define TT   2048
#define K_IN 256
#define NH   512
#define NG   2048   // 4*H
#define NC   64

typedef __bf16 bf16_t;
typedef bf16_t v16bf __attribute__((ext_vector_type(16)));
typedef bf16_t v8bf  __attribute__((ext_vector_type(8)));
typedef float  v8f   __attribute__((ext_vector_type(8)));
typedef unsigned int u32x4 __attribute__((ext_vector_type(4)));
typedef int          i32x4 __attribute__((ext_vector_type(4)));
typedef int          i32x8 __attribute__((ext_vector_type(8)));

// ---- scalar f32 <-> bf16 (RNE) on raw ushort storage ----
static __device__ __forceinline__ unsigned short f2bf(float x) {
  unsigned u = __builtin_bit_cast(unsigned, x);
  u += 0x7FFFu + ((u >> 16) & 1u);
  return (unsigned short)(u >> 16);
}
static __device__ __forceinline__ float bf2f(unsigned short h) {
  return __builtin_bit_cast(float, (unsigned)h << 16);
}

// ---- WMMA fragment loaders (contiguous k = 16*half + s order in both A and B, wave32) ----
// A (16x32 bf16): lane half = lane>>4; lane holds row m = lane&15,
//   elements 0..7  = K: half*16 .. half*16+7
//   elements 8..15 = K: half*16+8 .. half*16+15    (two contiguous 16B chunks)
static __device__ __forceinline__ v16bf load_a16(const unsigned short* rowbase,
                                                 int k0, int half) {
  v8bf lo = *(const v8bf*)(rowbase + k0 + half * 16);
  v8bf hi = *(const v8bf*)(rowbase + k0 + half * 16 + 8);
  return __builtin_shufflevector(lo, hi, 0, 1, 2, 3, 4, 5, 6, 7,
                                 8, 9, 10, 11, 12, 13, 14, 15);
}
// B (32x16 bf16): lane holds col n = lane&15; K = (lane>>4)*16 + 0..15
//   = one contiguous 32B chunk from the weight row (weights stored [N][K] row-major)
static __device__ __forceinline__ v16bf load_b16(const unsigned short* p) {
  return *(const v16bf*)p;
}

// =============== K0: f32 -> bf16 conversion ===============
__global__ void __launch_bounds__(256) cvt_bf16_kernel(const float* __restrict__ s,
                                                       unsigned short* __restrict__ d,
                                                       int n) {
  int i = blockIdx.x * 256 + threadIdx.x;
  if (i < n) d[i] = f2bf(s[i]);
}

// =============== K1: zero h_state + barrier counter ===============
__global__ void __launch_bounds__(256) init_kernel(unsigned short* __restrict__ hs,
                                                   unsigned int* __restrict__ bar) {
  int i = blockIdx.x * 256 + threadIdx.x;
  if (i < 2 * BB * NH) hs[i] = 0;      // both double buffers of h (h0 == 0)
  if (i == 0) *bar = 0;                 // grid-barrier counter (reset every launch)
}

// =============== K2: x_proj[t][b][g] = inputs @ W_ih^T + b_ih + b_hh (bf16 out) ===============
// Row r = t*32 + b (matches [T,B,4H] flattening). One wave per (M-tile, 64-col group).
__global__ void __launch_bounds__(256) xproj_gemm_kernel(
    const unsigned short* __restrict__ in_bf,   // [B][T][IN] bf16
    const unsigned short* __restrict__ wih,     // [4H][IN]  bf16
    const float* __restrict__ b_ih,
    const float* __restrict__ b_hh,
    unsigned short* __restrict__ xp)            // [T*B][4H] bf16
{
  const int tid  = threadIdx.x;
  const int wave = tid >> 5, lane = tid & 31;
  const int w  = blockIdx.x * 8 + wave;   // 131072 waves total
  const int mt = w >> 5;                  // 4096 M-tiles of 16 rows
  const int ng = w & 31;                  // 32 groups of 64 columns
  const int n = lane & 15, half = lane >> 4;

  const int r = mt * 16 + n;              // this lane's A row
  const int b = r & 31, t = r >> 5;
  const unsigned short* arow = in_bf + ((size_t)b * TT + t) * K_IN;
  const int g0 = ng * 64;

  v8f acc[4];
#pragma unroll
  for (int j = 0; j < 4; ++j) {
    float bias = b_ih[g0 + j * 16 + n] + b_hh[g0 + j * 16 + n];
#pragma unroll
    for (int v = 0; v < 8; ++v) acc[j][v] = bias;
  }

#pragma unroll
  for (int kc = 0; kc < K_IN / 32; ++kc) {
    v16bf a = load_a16(arow, kc * 32, half);
#pragma unroll
    for (int j = 0; j < 4; ++j) {
      const unsigned short* bp =
          wih + (size_t)(g0 + j * 16 + n) * K_IN + kc * 32 + half * 16;
      acc[j] = __builtin_amdgcn_wmma_f32_16x16x32_bf16(
          false, a, false, load_b16(bp), (short)0, acc[j], false, false);
    }
  }

  // C/D layout: element v -> row m = v + 8*half, col n
#pragma unroll
  for (int j = 0; j < 4; ++j) {
    int col = g0 + j * 16 + n;
#pragma unroll
    for (int v = 0; v < 8; ++v) {
      size_t rr = (size_t)(mt * 16 + v + 8 * half);
      xp[rr * NG + col] = f2bf(acc[j][v]);
    }
  }
}

// =============== K3: persistent LSTM scan (32 WGs, device-wide barrier/step) ===============
// WG = (batch-tile of 16) x (h-slice of 32). Each of 8 waves owns one 16x16 gate tile:
// gate gidx = wave>>1, columns gidx*512 + hbase + (wave&1)*16 + [0..16).
// W_hh slice lives in registers (loop invariant: 16 fragments = 128 VGPRs/wave).
// The 16x512 bf16 h-tile is staged into LDS once per WG per step by the Tensor
// Data Mover (tensor_load_to_lds), then read as ds_load A fragments by all waves.
__global__ void __launch_bounds__(256) lstm_scan_kernel(
    const unsigned short* __restrict__ xp,      // [T*B][4H] bf16
    const unsigned short* __restrict__ whh,     // [4H][H]   bf16
    const float* __restrict__ hnoise,           // [T][B][H] f32
    const float* __restrict__ cnoise,           // [T][B][H] f32
    unsigned short* __restrict__ hout,          // [B][T][H] bf16
    unsigned short* __restrict__ hstate,        // [2][B][H] bf16 (double buffer)
    unsigned int* __restrict__ bar)
{
  __shared__ __align__(32) unsigned short hstage[16 * NH];  // 16KB h tile
  __shared__ float gates[4][16][32];
  __shared__ float cst[16][32];

  const int tid  = threadIdx.x;
  const int wave = tid >> 5, lane = tid & 31;
  const int mtile = blockIdx.x & 1;           // which 16 batch rows
  const int hbase = (blockIdx.x >> 1) * 32;   // which 32 h-indices
  const int gidx = wave >> 1, sub = wave & 1;
  const int n = lane & 15, half = lane >> 4;
  const int col = gidx * NH + hbase + sub * 16 + n;   // gate column in [0,4H)

  // ---- Hoist this wave's W_hh slice into registers (invariant over t) ----
  const unsigned short* brow = whh + (size_t)col * NH + half * 16;
  v16bf bfr[NH / 32];
#pragma unroll
  for (int kc = 0; kc < NH / 32; ++kc) bfr[kc] = load_b16(brow + kc * 32);

  // ---- TDM descriptor (CDNA5 ISA 08_async_tensor 8.3/8.4): 2-D bf16 tile ----
  // group0: {count=1 | flags=0, lds_addr, global_addr[31:0], addr[56:32]|type=2<<30}
  // group1: word0: data_size=1 (2B); word1: tensor_dim0[15:0]<<16 (=512);
  //         word2: tensor_dim1[15:0]<<16 (=16);  word3: tile_dim0<<16 (=512);
  //         word4: tile_dim1 (=16); word5: tensor_dim0_stride (=512); rest 0.
  const unsigned lds_off = (unsigned)(size_t)&hstage[0];
  const size_t gtile0 = (size_t)hstate + (size_t)mtile * 16 * NH * 2;  // buf0 tile
  const size_t gtile1 = gtile0 + (size_t)BB * NH * 2;                  // buf1 tile
  const i32x8 g1d = { 0x00010000, (int)(512u << 16), (int)(16u << 16),
                      (int)(512u << 16), 16, 512, 0, 0 };
  const i32x4 z4 = { 0, 0, 0, 0 };
  const i32x8 z8 = { 0, 0, 0, 0, 0, 0, 0, 0 };

  for (int e = tid; e < 512; e += 256) cst[e >> 5][e & 31] = 0.f;
  __syncthreads();

#pragma unroll 1
  for (int t = 0; t < TT; ++t) {
    // ---- Stage h[t&1] tile into LDS with the Tensor Data Mover ----
    if (wave == 0) {
      size_t ga = (t & 1) ? gtile1 : gtile0;
      u32x4 g0d = { 1u, lds_off, (unsigned)ga,
                    (unsigned)((ga >> 32) & 0x01FFFFFFu) | (2u << 30) };
      __builtin_amdgcn_tensor_load_to_lds(g0d, g1d, z4, z4, z8, 0);
      __builtin_amdgcn_s_wait_tensorcnt(0);
    }
    __syncthreads();   // hstage now holds h for this step

    // Seed accumulator with precomputed x-projection (acts as the C matrix)
    v8f acc;
#pragma unroll
    for (int v = 0; v < 8; ++v) {
      int bg = mtile * 16 + v + 8 * half;
      acc[v] = bf2f(xp[((size_t)t * BB + bg) * NG + col]);
    }

    const unsigned short* arow = &hstage[n * NH];
#pragma unroll
    for (int kc = 0; kc < NH / 32; ++kc) {
      v16bf a = load_a16(arow, kc * 32, half);   // ds_load_b128 x2
      acc = __builtin_amdgcn_wmma_f32_16x16x32_bf16(
          false, a, false, bfr[kc], (short)0, acc, false, false);
    }

    if (t + 1 < TT)  // pull next step's x_proj toward the WGP
      __builtin_prefetch(xp + ((size_t)(t + 1) * BB + mtile * 16) * NG + col, 0, 1);

#pragma unroll
    for (int v = 0; v < 8; ++v)
      gates[gidx][v + 8 * half][sub * 16 + n] = acc[v];
    __syncthreads();

    // Elementwise cell update for this WG's 16 x 32 (batch x h) patch
    unsigned short* wb = hstate + (size_t)((t + 1) & 1) * (BB * NH);
    for (int e = tid; e < 512; e += 256) {
      int m = e >> 5, hh = e & 31;
      int bg = mtile * 16 + m;
      int hf = hbase + hh;
      float gi = gates[0][m][hh], gf = gates[1][m][hh];
      float gg = gates[2][m][hh], go = gates[3][m][hh];
      float iv = 1.f / (1.f + expf(-gi));
      float fv = 1.f / (1.f + expf(-gf));
      float ov = 1.f / (1.f + expf(-go));
      float gv = tanhf(gg);
      size_t noff = ((size_t)t * BB + bg) * NH + hf;
      float cn = fv * cst[m][hh] + iv * gv;
      float hn = ov * tanhf(cn) + hnoise[noff];
      cn += cnoise[noff];
      cst[m][hh] = cn;
      unsigned short hb = f2bf(hn);
      hout[((size_t)bg * TT + t) * NH + hf] = hb;
      wb[(size_t)bg * NH + hf] = hb;
    }

    // Device-wide barrier: writers of h[(t+1)&1] done before anyone reads it.
    // Release-add publishes stores; acquire spin-load invalidates WGP cache.
    __syncthreads();
    __threadfence();
    if (tid == 0) {
      __hip_atomic_fetch_add(bar, 1u, __ATOMIC_ACQ_REL, __HIP_MEMORY_SCOPE_AGENT);
      unsigned need = 32u * (unsigned)(t + 1);
      while (__hip_atomic_load(bar, __ATOMIC_ACQUIRE, __HIP_MEMORY_SCOPE_AGENT) < need)
        __builtin_amdgcn_s_sleep(2);
    }
    __syncthreads();
  }
}

// =============== K4: logits = h_out @ W_final^T + b_final, then log_softmax over C ===============
__global__ void __launch_bounds__(256) head_kernel(
    const unsigned short* __restrict__ hout,    // [B*T][H] bf16
    const unsigned short* __restrict__ wfin,    // [C][H]   bf16
    const float* __restrict__ b_final,
    float* __restrict__ out)                    // [B*T][C] f32
{
  __shared__ float tile[8][16][NC];
  const int tid = threadIdx.x, wave = tid >> 5, lane = tid & 31;
  const int n = lane & 15, half = lane >> 4;
  const int mt = blockIdx.x * 8 + wave;         // 4096 M-tiles
  const unsigned short* arow = hout + (size_t)(mt * 16 + n) * NH;

  v8f acc[4];
#pragma unroll
  for (int j = 0; j < 4; ++j) {
    float bias = b_final[j * 16 + n];
#pragma unroll
    for (int v = 0; v < 8; ++v) acc[j][v] = bias;
  }

#pragma unroll 4
  for (int kc = 0; kc < NH / 32; ++kc) {
    v16bf a = load_a16(arow, kc * 32, half);
#pragma unroll
    for (int j = 0; j < 4; ++j) {
      const unsigned short* bp =
          wfin + (size_t)(j * 16 + n) * NH + kc * 32 + half * 16;
      acc[j] = __builtin_amdgcn_wmma_f32_16x16x32_bf16(
          false, a, false, load_b16(bp), (short)0, acc[j], false, false);
    }
  }

#pragma unroll
  for (int j = 0; j < 4; ++j)
#pragma unroll
    for (int v = 0; v < 8; ++v)
      tile[wave][v + 8 * half][j * 16 + n] = acc[j][v];
  __syncthreads();

  if (lane < 16) {
    const float* rowp = tile[wave][lane];
    float mx = rowp[0];
#pragma unroll
    for (int c = 1; c < NC; ++c) mx = fmaxf(mx, rowp[c]);
    float s = 0.f;
#pragma unroll
    for (int c = 0; c < NC; ++c) s += expf(rowp[c] - mx);
    float lz = mx + logf(s);
    size_t r = (size_t)(mt * 16 + lane);
#pragma unroll
    for (int c = 0; c < NC; ++c) out[r * NC + c] = rowp[c] - lz;
  }
}

// ======================= launch =======================
extern "C" void kernel_launch(void* const* d_in, const int* in_sizes, int n_in,
                              void* d_out, int out_size, void* d_ws, size_t ws_size,
                              hipStream_t stream) {
  (void)in_sizes; (void)n_in; (void)out_size; (void)ws_size;
  const float* inputs  = (const float*)d_in[0];   // [B,T,IN]
  const float* hnoise  = (const float*)d_in[1];   // [T,B,H]
  const float* cnoise  = (const float*)d_in[2];   // [T,B,H]
  const float* W_ih    = (const float*)d_in[3];   // [4H,IN]
  const float* W_hh    = (const float*)d_in[4];   // [4H,H]
  const float* b_ih    = (const float*)d_in[5];
  const float* b_hh    = (const float*)d_in[6];
  const float* W_final = (const float*)d_in[7];   // [C,H]
  const float* b_final = (const float*)d_in[8];
  float* out = (float*)d_out;

  char* ws = (char*)d_ws;
  size_t off = 0;
  auto carve = [&](size_t bytes) {
    void* p = ws + off;
    off = (off + bytes + 255) & ~(size_t)255;
    return p;
  };
  unsigned short* in_bf  = (unsigned short*)carve((size_t)BB * TT * K_IN * 2); //  32MB
  unsigned short* wih_bf = (unsigned short*)carve((size_t)NG * K_IN * 2);      //   1MB
  unsigned short* whh_bf = (unsigned short*)carve((size_t)NG * NH * 2);        //   2MB
  unsigned short* wfn_bf = (unsigned short*)carve((size_t)NC * NH * 2);        //  64KB
  unsigned short* xp_bf  = (unsigned short*)carve((size_t)TT * BB * NG * 2);   // 256MB
  unsigned short* hout   = (unsigned short*)carve((size_t)BB * TT * NH * 2);   //  64MB
  unsigned short* hstate = (unsigned short*)carve((size_t)2 * BB * NH * 2);    //  64KB
  unsigned int*   bar    = (unsigned int*)carve(256);

  // f32 -> bf16 conversions
  cvt_bf16_kernel<<<(BB * TT * K_IN + 255) / 256, 256, 0, stream>>>(inputs, in_bf, BB * TT * K_IN);
  cvt_bf16_kernel<<<(NG * K_IN + 255) / 256, 256, 0, stream>>>(W_ih, wih_bf, NG * K_IN);
  cvt_bf16_kernel<<<(NG * NH + 255) / 256, 256, 0, stream>>>(W_hh, whh_bf, NG * NH);
  cvt_bf16_kernel<<<(NC * NH + 255) / 256, 256, 0, stream>>>(W_final, wfn_bf, NC * NH);
  init_kernel<<<(2 * BB * NH + 255) / 256, 256, 0, stream>>>(hstate, bar);

  // Parallel phase: input projection GEMM (65536 x 2048 x 256)
  xproj_gemm_kernel<<<16384, 256, 0, stream>>>(in_bf, wih_bf, b_ih, b_hh, xp_bf);

  // Serial phase: persistent 32-WG scan with one device barrier per step
  lstm_scan_kernel<<<32, 256, 0, stream>>>(xp_bf, whh_bf, hnoise, cnoise, hout, hstate, bar);

  // Head GEMM + log_softmax (65536 x 64 x 512)
  head_kernel<<<512, 256, 0, stream>>>(hout, wfn_bf, b_final, out);
}
